// GraphSAGE_61409442398712
// MI455X (gfx1250) — matrix operand order repaired and checked by reading, not compiled, not run on hardware
//
#include <hip/hip_runtime.h>
#include <hip/hip_bf16.h>
#include <math.h>

// ---------------------------------------------------------------------------
// GraphSAGE (3 layers, mean aggregation) for MI455X / gfx1250, wave32.
//   per layer: agg = scatter_mean(x[src] -> dst);  h = agg@Wl + bl + x@Wr
//   GEMM fused as [agg | x] (K=256) @ [Wl ; Wr] via v_wmma_f32_16x16x32_bf16
// ---------------------------------------------------------------------------

typedef __bf16 bf16_t;
typedef bf16_t v16bf __attribute__((ext_vector_type(16)));
typedef bf16_t v2bf  __attribute__((ext_vector_type(2)));
typedef float  v8f   __attribute__((ext_vector_type(8)));

union ABFrag { unsigned u[8]; v16bf v; };

static constexpr int NN = 50000;   // nodes
static constexpr int NE = 800000;  // edges
static constexpr int FD = 128;     // feature dim (F_IN == HIDDEN)
static constexpr int NG = 256;     // graphs
static constexpr int NC = 10;      // classes
static constexpr int KTOT = 256;   // fused K = 128 (agg) + 128 (self)

// hardware packed f32->bf16 convert (v_cvt_pk_bf16_f32 class), RNE by default
__device__ __forceinline__ unsigned pack2(float a, float b) {
    v2bf p;
    p.x = (bf16_t)a;
    p.y = (bf16_t)b;
    return __builtin_bit_cast(unsigned, p);
}

// ---------------------------------------------------------------------------
__global__ __launch_bounds__(256) void zero_f32(float* __restrict__ p, int n) {
    int i = blockIdx.x * 256 + threadIdx.x;
    if (i < n) p[i] = 0.0f;
}

// one wave32 per edge: 4 coalesced f32 atomics per lane (128 feats)
__global__ __launch_bounds__(256) void scatter_edges(
    const int* __restrict__ src, const int* __restrict__ dst,
    const float* __restrict__ feat, float* __restrict__ ssum,
    float* __restrict__ cnt) {
    int tid  = blockIdx.x * 256 + threadIdx.x;
    int e    = tid >> 5;
    int lane = tid & 31;
    if (e >= NE) return;
    int s = src[e], d = dst[e];
    const float* f = feat + (long)s * FD;
    float*       o = ssum + (long)d * FD;
#pragma unroll
    for (int j = 0; j < 4; ++j)
        atomicAdd(&o[lane + 32 * j], f[lane + 32 * j]);
    if (lane == 0) atomicAdd(&cnt[d], 1.0f);
}

__global__ __launch_bounds__(256) void invert_cnt(float* __restrict__ cnt, int n) {
    int i = blockIdx.x * 256 + threadIdx.x;
    if (i < n) cnt[i] = 1.0f / fmaxf(cnt[i], 1.0f);
}

// Pre-swizzle [Wl ; Wr] (K=256 x nreal) into WMMA bf16 B-fragment layout.
// dword id = ((ct*8 + ks)*32 + lane)*8 + v ;  B: n = ct*16+lane%16,
// k = ks*32 + (lane/16)*8 + (v/4)*16 + (v%4)*2 + {0,1}   (ISA 7.12.2, 16-bit B 32x16)
__global__ __launch_bounds__(256) void pack_weights(
    const float* __restrict__ Wl, const float* __restrict__ Wr,
    unsigned* __restrict__ Wp, int nout_pad, int nreal) {
    int id = blockIdx.x * 256 + threadIdx.x;
    int total = (nout_pad / 16) * 8 * 32 * 8;
    if (id >= total) return;
    int v    = id & 7;
    int lane = (id >> 3) & 31;
    int ks   = (id >> 8) & 7;
    int ct   = id >> 11;
    int n = ct * 16 + (lane & 15);
    int k = ks * 32 + (lane >> 4) * 8 + ((v >> 2) << 4) + ((v & 3) << 1);
    float f0 = 0.0f, f1 = 0.0f;
    if (n < nreal) {
        f0 = (k     < 128) ? Wl[k * nreal + n]         : Wr[(k - 128) * nreal + n];
        f1 = (k + 1 < 128) ? Wl[(k + 1) * nreal + n]   : Wr[(k - 127) * nreal + n];
    }
    Wp[id] = pack2(f0, f1);
}

// Fused SAGE GEMM: out[r, 0:nreal] = (agg[r]*icnt[r]) @ Wl + prev[r] @ Wr + b
// block = 128 threads = 4 waves; each wave: 16 rows x NOUT cols.
template <int NOUT>
__global__ __launch_bounds__(128) void sage_gemm_wmma(
    const float* __restrict__ agg, const float* __restrict__ icnt,
    const float* __restrict__ prev, const unsigned* __restrict__ Wp,
    const float* __restrict__ bias, float* __restrict__ out, int nreal) {
    constexpr int NT = NOUT / 16;
    const int lane    = threadIdx.x & 31;
    const int wave    = threadIdx.x >> 5;
    const int rowBase = blockIdx.x * 64 + wave * 16;
    const int m = lane & 15;
    const int h = lane >> 4;

    int rowA = rowBase + m;
    if (rowA > NN - 1) rowA = NN - 1;          // clamp: keep EXEC all-ones
    const long rowOff = (long)rowA * FD;
    const float inv   = icnt[rowA];

    v8f acc[NT] = {};
    ABFrag a, b;

#pragma unroll
    for (int ks = 0; ks < KTOT / 32; ++ks) {
        // A fragment (ISA 7.12.2, 16-bit A 16x32): lane half h, per-VGPR pairs
        const int kb = ks * 32 + h * 8;
#pragma unroll
        for (int v = 0; v < 8; ++v) {
            int k = kb + ((v >> 2) << 4) + ((v & 3) << 1);
            float f0, f1;
            if (ks < 4) {                       // K 0..127  -> mean-aggregated
                f0 = agg[rowOff + k] * inv;
                f1 = agg[rowOff + k + 1] * inv;
            } else {                            // K 128..255 -> self features
                f0 = prev[rowOff + k - 128];
                f1 = prev[rowOff + k - 127];
            }
            a.u[v] = pack2(f0, f1);
        }
#pragma unroll
        for (int t = 0; t < NT; ++t) {
            const uint4* bp =
                (const uint4*)(Wp + ((((t * 8 + ks) * 32) + lane) << 3));
            uint4 lo = bp[0], hi = bp[1];
            b.u[0] = lo.x; b.u[1] = lo.y; b.u[2] = lo.z; b.u[3] = lo.w;
            b.u[4] = hi.x; b.u[5] = hi.y; b.u[6] = hi.z; b.u[7] = hi.w;
            acc[t] = __builtin_amdgcn_wmma_f32_16x16x32_bf16(
                false, a.v, false, b.v, (short)0, acc[t], false, false);
        }
    }

    // C/D layout: VGPR i -> row = i + h*8, col = lane%16
#pragma unroll
    for (int t = 0; t < NT; ++t) {
        int col = t * 16 + m;
        float bv = (col < nreal) ? bias[col] : 0.0f;
#pragma unroll
        for (int i = 0; i < 8; ++i) {
            int r = rowBase + i + h * 8;
            if (r < NN && col < nreal)
                out[(long)r * NOUT + col] = acc[t][i] + bv;
        }
    }
}

// global mean pool: per-node atomic adds into [NG x 16] sums
__global__ __launch_bounds__(256) void pool_nodes(
    const float* __restrict__ h3, const int* __restrict__ batch,
    float* __restrict__ gsum, float* __restrict__ gcnt) {
    int nidx = blockIdx.x * 256 + threadIdx.x;
    if (nidx >= NN) return;
    int g = batch[nidx];
#pragma unroll
    for (int c = 0; c < NC; ++c)
        atomicAdd(&gsum[g * 16 + c], h3[(long)nidx * 16 + c]);
    atomicAdd(&gcnt[g], 1.0f);
}

__global__ __launch_bounds__(256) void pool_finalize(
    const float* __restrict__ gsum, const float* __restrict__ gcnt,
    float* __restrict__ out) {
    int g = blockIdx.x * 256 + threadIdx.x;
    if (g >= NG) return;
    float inv = 1.0f / fmaxf(gcnt[g], 1.0f);
    float v[NC], mx = -INFINITY;
#pragma unroll
    for (int c = 0; c < NC; ++c) {
        v[c] = gsum[g * 16 + c] * inv;
        mx = fmaxf(mx, v[c]);
    }
    float se = 0.0f;
#pragma unroll
    for (int c = 0; c < NC; ++c) se += expf(v[c] - mx);
    float lse = mx + logf(se);
#pragma unroll
    for (int c = 0; c < NC; ++c) out[g * NC + c] = v[c] - lse;
}

// ---------------------------------------------------------------------------
extern "C" void kernel_launch(void* const* d_in, const int* in_sizes, int n_in,
                              void* d_out, int out_size, void* d_ws, size_t ws_size,
                              hipStream_t stream) {
    const float* x   = (const float*)d_in[0];
    const int*   ei  = (const int*)d_in[1];
    const int*   src = ei;
    const int*   dst = ei + NE;
    const int*   batch = (const int*)d_in[2];
    const float* Wl1 = (const float*)d_in[3];
    const float* bl1 = (const float*)d_in[4];
    const float* Wr1 = (const float*)d_in[5];
    const float* Wl2 = (const float*)d_in[6];
    const float* bl2 = (const float*)d_in[7];
    const float* Wr2 = (const float*)d_in[8];
    const float* Wl3 = (const float*)d_in[9];
    const float* bl3 = (const float*)d_in[10];
    const float* Wr3 = (const float*)d_in[11];
    float* out = (float*)d_out;

    // workspace carve-out (256-byte aligned)
    char* ws = (char*)d_ws;
    size_t off = 0;
    auto take = [&](size_t bytes) {
        char* p = ws + off;
        off += (bytes + 255) & ~(size_t)255;
        return p;
    };
    float*    agg  = (float*)take((size_t)NN * FD * 4);
    float*    cnt  = (float*)take((size_t)NN * 4);
    float*    hA   = (float*)take((size_t)NN * FD * 4);
    float*    hB   = (float*)take((size_t)NN * FD * 4);
    float*    h3   = (float*)take((size_t)NN * 16 * 4);
    unsigned* Wp1  = (unsigned*)take(8 * 8 * 32 * 8 * 4);   // 64 KB
    unsigned* Wp2  = (unsigned*)take(8 * 8 * 32 * 8 * 4);
    unsigned* Wp3  = (unsigned*)take(1 * 8 * 32 * 8 * 4);   // 8 KB
    float*    gsum = (float*)take((size_t)NG * 16 * 4);
    float*    gcnt = (float*)take((size_t)NG * 4);

    const int gemmGrid = (NN + 63) / 64;                    // 782
    const int edgeGrid = (NE * 32 + 255) / 256;             // 100000

    // pre-swizzle weights into WMMA B-fragment layout (bf16)
    pack_weights<<<(16384 + 255) / 256, 256, 0, stream>>>(Wl1, Wr1, Wp1, 128, 128);
    pack_weights<<<(16384 + 255) / 256, 256, 0, stream>>>(Wl2, Wr2, Wp2, 128, 128);
    pack_weights<<<(2048 + 255) / 256, 256, 0, stream>>>(Wl3, Wr3, Wp3, 16, NC);

    // ---- layer 1 ----
    zero_f32<<<(NN * FD + 255) / 256, 256, 0, stream>>>(agg, NN * FD);
    zero_f32<<<(NN + 255) / 256, 256, 0, stream>>>(cnt, NN);
    scatter_edges<<<edgeGrid, 256, 0, stream>>>(src, dst, x, agg, cnt);
    invert_cnt<<<(NN + 255) / 256, 256, 0, stream>>>(cnt, NN);
    sage_gemm_wmma<128><<<gemmGrid, 128, 0, stream>>>(agg, cnt, x, Wp1, bl1, hA, 128);

    // ---- layer 2 ----
    zero_f32<<<(NN * FD + 255) / 256, 256, 0, stream>>>(agg, NN * FD);
    zero_f32<<<(NN + 255) / 256, 256, 0, stream>>>(cnt, NN);
    scatter_edges<<<edgeGrid, 256, 0, stream>>>(src, dst, hA, agg, cnt);
    invert_cnt<<<(NN + 255) / 256, 256, 0, stream>>>(cnt, NN);
    sage_gemm_wmma<128><<<gemmGrid, 128, 0, stream>>>(agg, cnt, hA, Wp2, bl2, hB, 128);

    // ---- layer 3 ----
    zero_f32<<<(NN * FD + 255) / 256, 256, 0, stream>>>(agg, NN * FD);
    zero_f32<<<(NN + 255) / 256, 256, 0, stream>>>(cnt, NN);
    scatter_edges<<<edgeGrid, 256, 0, stream>>>(src, dst, hB, agg, cnt);
    invert_cnt<<<(NN + 255) / 256, 256, 0, stream>>>(cnt, NN);
    sage_gemm_wmma<16><<<gemmGrid, 128, 0, stream>>>(agg, cnt, hB, Wp3, bl3, h3, NC);

    // ---- global mean pool + log_softmax ----
    zero_f32<<<(NG * 16 + 255) / 256, 256, 0, stream>>>(gsum, NG * 16);
    zero_f32<<<(NG + 255) / 256, 256, 0, stream>>>(gcnt, NG);
    pool_nodes<<<(NN + 255) / 256, 256, 0, stream>>>(h3, batch, gsum, gcnt);
    pool_finalize<<<(NG + 255) / 256, 256, 0, stream>>>(gsum, gcnt, out);
}
